// SelfAttention_4973572129200
// MI455X (gfx1250) — matrix operand order repaired and checked by reading, not compiled, not run on hardware
//
#include <hip/hip_runtime.h>
#include <hip/hip_bf16.h>

// ---------------------------------------------------------------------------
// Self-attention (B=8, S=2048, D=1024, H=64), fp32 in/out, bf16 WMMA compute.
// Kernel 0: one-shot fp32->bf16 weight conversion (if workspace allows)
// Kernel 1: QKV projection  -> workspace (q,k row-major bf16; v transposed)
// Kernel 2: causal flash attention with online softmax, WMMA QK^T and P*V,
//           K/V tiles staged in LDS via GLOBAL_LOAD_ASYNC_TO_LDS (ASYNCcnt).
// ---------------------------------------------------------------------------

typedef __bf16 bf16;
typedef __attribute__((ext_vector_type(16))) __bf16 v16bf;
typedef __attribute__((ext_vector_type(8)))  __bf16 v8bf;
typedef __attribute__((ext_vector_type(8)))  float  v8f;
typedef __attribute__((ext_vector_type(4)))  int    v4i;

union V16U { v16bf v; v8bf h[2]; };

static constexpr int BB = 8;
static constexpr int SS = 2048;
static constexpr int DD = 1024;
static constexpr int HH = 64;
static constexpr int MTOT = BB * SS;   // 16384

#if defined(__has_builtin)
#  if __has_builtin(__builtin_amdgcn_global_load_async_to_lds_b128)
#    define HAVE_ASYNC_LDS 1
#  endif
#endif
#ifndef HAVE_ASYNC_LDS
#  define HAVE_ASYNC_LDS 0
#endif

// exact-typed address-space casts for the async builtin (param type is
// "int __vector(4) __device__ *" per clang diagnostics)
#define ASYNC_GPTR(p) ((__attribute__((address_space(1))) v4i*)(p))
#define ASYNC_LPTR(p) ((__attribute__((address_space(3))) v4i*)(p))

__device__ __forceinline__ v8f wmma_bf(v16bf a, v16bf b, v8f c) {
  // D = A(16x32 bf16) * B(32x16 bf16) + C(16x16 f32)
  return __builtin_amdgcn_wmma_f32_16x16x32_bf16(
      /*neg_a=*/false, a, /*neg_b=*/false, b,
      /*c_mod=*/(short)0, c, /*reuse_a=*/false, /*reuse_b=*/false);
}

// A fragment (16x32 bf16) from fp32 row-major source, with convert.
// lane m = lane&15 holds rows; per lane one contiguous 16-elem run:
//   K = hi*16 .. +16   (matches contiguous B fragment order)
__device__ __forceinline__ v16bf a_frag_f32(const float* base, int ld, int lane, int kc) {
  const int m = lane & 15, hi = lane >> 4;
  const float* p0 = base + (size_t)m * ld + kc + hi * 16;
  const float* p1 = p0 + 8;
  v8f a0 = *(const v8f*)p0;
  v8f a1 = *(const v8f*)p1;
  v16bf r;
#pragma unroll
  for (int e = 0; e < 8; ++e) { r[e] = (__bf16)a0[e]; r[8 + e] = (__bf16)a1[e]; }
  return r;
}

// A fragment (16x32 bf16) from bf16 row-major source (global or LDS).
__device__ __forceinline__ v16bf a_frag_bf(const bf16* base, int ld, int lane, int kc) {
  const int m = lane & 15, hi = lane >> 4;
  const bf16* p0 = base + (size_t)m * ld + kc + hi * 16;
  V16U u;
  u.h[0] = *(const v8bf*)p0;
  u.h[1] = *(const v8bf*)(p0 + 8);
  return u.v;
}

// B fragment (32x16 bf16): lane holds column n = lane&15, K rows hi*16..+16
// contiguous.  `base` = row-major [col][K] storage with stride ld.
__device__ __forceinline__ v16bf b_frag_bf(const bf16* base, int ld, int lane, int kc) {
  const int n = lane & 15, hi = lane >> 4;
  const bf16* p = base + (size_t)n * ld + kc + hi * 16;
  V16U u;
  u.h[0] = *(const v8bf*)p;
  u.h[1] = *(const v8bf*)(p + 8);
  return u.v;
}

// B fragment (32x16 bf16) from fp32 weight W[h][d] (row-major, ld=1024):
// B[d][h] = W[h][d]  ->  lane reads W[nt*16+n][kc + hi*16 .. +16] contiguous.
__device__ __forceinline__ v16bf b_frag_w(const float* W, int lane, int kc, int nt) {
  const int n = lane & 15, hi = lane >> 4;
  const float* p = W + (size_t)(nt * 16 + n) * DD + kc + hi * 16;
  v8f b0 = *(const v8f*)p;
  v8f b1 = *(const v8f*)(p + 8);
  v16bf r;
#pragma unroll
  for (int e = 0; e < 8; ++e) { r[e] = (__bf16)b0[e]; r[8 + e] = (__bf16)b1[e]; }
  return r;
}

// ---------------------------------------------------------------------------
// Kernel 0: convert the three 64x1024 fp32 weights to bf16 (once).
// ---------------------------------------------------------------------------
__global__ __launch_bounds__(256) void wcvt_kernel(
    const float* __restrict__ Wq, const float* __restrict__ Wk,
    const float* __restrict__ Wv, bf16* __restrict__ out) {
  const int i = blockIdx.x * 256 + threadIdx.x;      // 49152 threads, 4 elems each
  const float* src[3] = {Wq, Wk, Wv};
  const int w = i / 16384;
  const int e = (i - w * 16384) * 4;
  const float4 f = *(const float4*)(src[w] + e);
  bf16* o = out + (size_t)w * (HH * DD) + e;
  o[0] = (bf16)f.x; o[1] = (bf16)f.y; o[2] = (bf16)f.z; o[3] = (bf16)f.w;
}

// ---------------------------------------------------------------------------
// Kernel 1: q = x Wq^T, k = x Wk^T, v = x Wv^T  (M=16384, N=64, K=1024)
// grid 128 x 256thr; wave handles 16 rows, 12 WMMA per 32-wide K chunk.
// WBF=true: B fragments load pre-converted bf16 weights (no cvt in loop).
// ---------------------------------------------------------------------------
template <bool WBF>
__global__ __launch_bounds__(256) void qkv_kernel(
    const float* __restrict__ x,
    const float* __restrict__ Wq, const float* __restrict__ Wk,
    const float* __restrict__ Wv,
    const bf16* __restrict__ Wqb, const bf16* __restrict__ Wkb,
    const bf16* __restrict__ Wvb,
    bf16* __restrict__ qo, bf16* __restrict__ ko, bf16* __restrict__ vT) {
  const int lane = threadIdx.x & 31;
  const int wv   = threadIdx.x >> 5;
  const int m0   = blockIdx.x * 128 + wv * 16;
  const float* xrow = x + (size_t)m0 * DD;

  v8f acc[3][4] = {};
  const float* Wsf[3] = {Wq, Wk, Wv};
  const bf16*  Wsb[3] = {Wqb, Wkb, Wvb};

  for (int kc = 0; kc < DD; kc += 32) {
    const v16bf a = a_frag_f32(xrow, DD, lane, kc);
#pragma unroll
    for (int wgt = 0; wgt < 3; ++wgt) {
#pragma unroll
      for (int nt = 0; nt < 4; ++nt) {
        v16bf bf;
        if (WBF) bf = b_frag_bf(Wsb[wgt] + (size_t)nt * 16 * DD, DD, lane, kc);
        else     bf = b_frag_w(Wsf[wgt], lane, kc, nt);
        acc[wgt][nt] = wmma_bf(a, bf, acc[wgt][nt]);
      }
    }
  }

  // epilogue: C layout lane(n=lane&15, hi=lane>>4), VGPR r -> (M=r+8*hi, N=n)
  const int n = lane & 15, hi = lane >> 4;
#pragma unroll
  for (int nt = 0; nt < 4; ++nt) {
#pragma unroll
    for (int r = 0; r < 8; ++r) {
      const int m   = m0 + r + 8 * hi;
      const int col = nt * 16 + n;
      qo[(size_t)m * HH + col] = (bf16)acc[0][nt][r];
      ko[(size_t)m * HH + col] = (bf16)acc[1][nt][r];
      const int bb = m >> 11, s = m & (SS - 1);
      vT[((size_t)bb * HH + col) * SS + s] = (bf16)acc[2][nt][r];
    }
  }
}

// ---------------------------------------------------------------------------
// Kernel 2: causal flash attention.  grid (32,8) x 128thr.
// Each wave: one 16x64 Q tile; WG streams 64-key K/V tiles through LDS
// using async global->LDS copies when available.
// ---------------------------------------------------------------------------
__global__ __launch_bounds__(128) void attn_kernel(
    const bf16* __restrict__ q, const bf16* __restrict__ k,
    const bf16* __restrict__ vT, float* __restrict__ out) {
  __shared__ __align__(16) bf16 Klds[64][64];     // [key][d]
  __shared__ __align__(16) bf16 Vtlds[64][64];    // [d][key]
  __shared__ __align__(16) float Slds[4][16][64]; // per-wave scores
  __shared__ __align__(16) bf16  Plds[4][16][64]; // per-wave probabilities
  __shared__ float rowscale[4][16];
  __shared__ float rowsum[4][16];

  const int tid  = threadIdx.x;
  const int lane = tid & 31;
  const int wv   = tid >> 5;
  const int qb   = blockIdx.x;          // query block (64 rows)
  const int bb   = blockIdx.y;          // batch
  const int qm0  = qb * 64 + wv * 16;   // wave's q rows (within batch)
  const size_t qbase = ((size_t)bb * SS + qm0) * HH;

  // Q fragments held in registers for the whole kernel (hdim 64 = 2 chunks)
  v16bf aq0 = a_frag_bf(q + qbase, HH, lane, 0);
  v16bf aq1 = a_frag_bf(q + qbase, HH, lane, 32);

  v8f o[4] = {};
  float m_run = -1e30f, l_run = 0.0f;
  const int n = lane & 15, hi = lane >> 4;
  const int row2 = lane >> 1, hh = lane & 1;  // softmax: 2 lanes per row

  const int nkb = qb + 1;  // causal: key blocks 0..qb
  for (int kb = 0; kb < nkb; ++kb) {
    __syncthreads();  // protect LDS tiles from previous iteration
    // cooperative staging: K tile row-major, V tile transposed ([d][key])
    for (int c = tid; c < 512; c += 128) {
      const int rrow = c >> 3, c8 = (c & 7) * 8;
      const bf16* gK = &k[((size_t)bb * SS + kb * 64 + rrow) * HH + c8];
      const bf16* gV = &vT[((size_t)bb * HH + rrow) * SS + kb * 64 + c8];
      bf16* lK = &Klds[rrow][c8];
      bf16* lV = &Vtlds[rrow][c8];
#if HAVE_ASYNC_LDS
      __builtin_amdgcn_global_load_async_to_lds_b128(ASYNC_GPTR(gK), ASYNC_LPTR(lK), 0, 0);
      __builtin_amdgcn_global_load_async_to_lds_b128(ASYNC_GPTR(gV), ASYNC_LPTR(lV), 0, 0);
#else
      *(v8bf*)lK = *(const v8bf*)gK;
      *(v8bf*)lV = *(const v8bf*)gV;
#endif
    }
    if (kb + 1 < nkb) {  // prefetch next tile toward L2/L0
      __builtin_prefetch(&k[((size_t)bb * SS + (kb + 1) * 64) * HH + lane * 16], 0, 0);
      __builtin_prefetch(&vT[(size_t)bb * HH * SS + (kb + 1) * 64 + (lane & 7) * 8], 0, 0);
    }
    asm volatile("s_wait_asynccnt 0x0" ::: "memory");  // drain async->LDS
    __syncthreads();

    // ---- scores: S = Q * K^T  (4 tiles of 16 keys, K-dim 64 = 2 WMMA each)
#pragma unroll
    for (int nt = 0; nt < 4; ++nt) {
      v8f s = {};
      s = wmma_bf(aq0, b_frag_bf(&Klds[nt * 16][0], 64, lane, 0), s);
      s = wmma_bf(aq1, b_frag_bf(&Klds[nt * 16][0], 64, lane, 32), s);
#pragma unroll
      for (int r = 0; r < 8; ++r) {
        const int qrow = qm0 + r + 8 * hi;
        const int key  = kb * 64 + nt * 16 + n;
        Slds[wv][r + 8 * hi][nt * 16 + n] =
            (key <= qrow) ? s[r] * 0.125f : -1e30f;  // 1/sqrt(64) fused
      }
    }
    __syncthreads();

    // ---- online softmax: each lane-pair owns one q row (32 keys each)
    float lm = -1e30f;
#pragma unroll
    for (int j = 0; j < 32; ++j) lm = fmaxf(lm, Slds[wv][row2][hh * 32 + j]);
    lm = fmaxf(lm, __shfl_xor(lm, 1, 32));
    const float m_new = fmaxf(m_run, lm);
    const float corr  = __expf(m_run - m_new);
    float sum = 0.0f;
#pragma unroll
    for (int j = 0; j < 32; ++j) {
      const float p = __expf(Slds[wv][row2][hh * 32 + j] - m_new);
      Plds[wv][row2][hh * 32 + j] = (bf16)p;
      sum += p;
    }
    sum += __shfl_xor(sum, 1, 32);
    l_run = l_run * corr + sum;
    m_run = m_new;
    if (hh == 0) rowscale[wv][row2] = corr;
    __syncthreads();

    // ---- rescale running output by row correction factor
#pragma unroll
    for (int r = 0; r < 8; ++r) {
      const float cr = rowscale[wv][r + 8 * hi];
#pragma unroll
      for (int nt = 0; nt < 4; ++nt) o[nt][r] *= cr;
    }

    // ---- O += P * V   (A = P 16x64, B = V 64x16 tiles via Vt layout)
    v16bf ap0 = a_frag_bf(&Plds[wv][0][0], 64, lane, 0);
    v16bf ap1 = a_frag_bf(&Plds[wv][0][0], 64, lane, 32);
#pragma unroll
    for (int nt = 0; nt < 4; ++nt) {
      o[nt] = wmma_bf(ap0, b_frag_bf(&Vtlds[nt * 16][0], 64, lane, 0), o[nt]);
      o[nt] = wmma_bf(ap1, b_frag_bf(&Vtlds[nt * 16][0], 64, lane, 32), o[nt]);
    }
  }

  // ---- epilogue: divide by softmax denominator, write fp32 output
  if (hh == 0) rowsum[wv][row2] = l_run;
  __syncthreads();
#pragma unroll
  for (int r = 0; r < 8; ++r) {
    const float inv = 1.0f / rowsum[wv][r + 8 * hi];
    const int qrow  = qm0 + r + 8 * hi;
    const size_t ob = ((size_t)bb * SS + qrow) * HH;
#pragma unroll
    for (int nt = 0; nt < 4; ++nt) out[ob + nt * 16 + n] = o[nt][r] * inv;
  }
}

// ---------------------------------------------------------------------------
extern "C" void kernel_launch(void* const* d_in, const int* in_sizes, int n_in,
                              void* d_out, int out_size, void* d_ws, size_t ws_size,
                              hipStream_t stream) {
  const float* x  = (const float*)d_in[0];
  const float* Wq = (const float*)d_in[1];
  const float* Wk = (const float*)d_in[2];
  const float* Wv = (const float*)d_in[3];

  bf16* qws  = (bf16*)d_ws;                       // [16384][64] bf16
  bf16* kws  = qws + (size_t)MTOT * HH;           // [16384][64] bf16
  bf16* vTws = kws + (size_t)MTOT * HH;           // [8][64][2048] bf16
  bf16* wbuf = vTws + (size_t)MTOT * HH;          // [3][64][1024] bf16

  const size_t need_elems = (size_t)3 * MTOT * HH + (size_t)3 * HH * DD;
  const bool cvt_weights = ws_size >= need_elems * sizeof(bf16);

  if (cvt_weights) {
    wcvt_kernel<<<192, 256, 0, stream>>>(Wq, Wk, Wv, wbuf);
    qkv_kernel<true><<<MTOT / 128, 256, 0, stream>>>(
        x, Wq, Wk, Wv, wbuf, wbuf + (size_t)HH * DD, wbuf + (size_t)2 * HH * DD,
        qws, kws, vTws);
  } else {
    qkv_kernel<false><<<MTOT / 128, 256, 0, stream>>>(
        x, Wq, Wk, Wv, nullptr, nullptr, nullptr, qws, kws, vTws);
  }
  attn_kernel<<<dim3(SS / 64, BB), 128, 0, stream>>>(qws, kws, vTws, (float*)d_out);
}
